// SingleHeadAttention_40097814675860
// MI455X (gfx1250) — matrix operand order repaired and checked by reading, not compiled, not run on hardware
//
#include <hip/hip_runtime.h>
#include <hip/hip_bf16.h>

#define S_LEN  4096
#define H_DIM  768
#define D_DIM  64
#define BATCH  8
#define KSTEPS 24   // H_DIM / 32

typedef __attribute__((ext_vector_type(16))) __bf16 v16bf;
typedef __attribute__((ext_vector_type(8)))  __bf16 v8bf;
typedef __attribute__((ext_vector_type(8)))  float  v8f;

union V16 { v16bf v; v8bf h[2]; };
union V8  { v8bf  v; __bf16 e[8]; };

static __device__ __forceinline__ v8f wmma_bf16(v16bf a, v16bf b, v8f c) {
  return __builtin_amdgcn_wmma_f32_16x16x32_bf16(false, a, false, b, (short)0, c,
                                                 false, false);
}

// ---------------------------------------------------------------------------
// Phase 0: swizzle Wq/Wk/Wv (fp32, row-major [768,64]) into the exact bf16
// B-operand layout: tile index ((mat*24+ks)*4+nt), per-lane 16 bf16 where
// lane n = nt*16 + (lane&15), k = ks*32 + (lane>>4)*8 + (i&7) + (i>>3)*16 (table order).
// ---------------------------------------------------------------------------
__global__ void wprep_kernel(const float* __restrict__ Wq,
                             const float* __restrict__ Wk,
                             const float* __restrict__ Wv,
                             __bf16* __restrict__ wswz) {
  const int lane = threadIdx.x;          // 32 threads/block
  const int tile = blockIdx.x;           // 3*24*4 = 288 tiles
  const int nt   = tile & 3;
  const int ks   = (tile >> 2) % KSTEPS;
  const int mat  = tile / (KSTEPS * 4);
  const float* W = (mat == 0) ? Wq : ((mat == 1) ? Wk : Wv);
  const int n    = nt * 16 + (lane & 15);
  const int half = lane >> 4;
  V16 o;
#pragma unroll
  for (int i = 0; i < 16; ++i) {
    const int kk = ks * 32 + half * 8 + (i & 7) + (i >> 3) * 16;
    o.v[i] = (__bf16)W[(size_t)kk * D_DIM + n];
  }
  __bf16* dst = wswz + ((size_t)tile * 32 + lane) * 16;
  *(v8bf*)dst       = o.h[0];
  *(v8bf*)(dst + 8) = o.h[1];
}

// ---------------------------------------------------------------------------
// Phase 1: fused QKV projection, software-pipelined. One wave = 16 rows of x,
// all 192 output columns (12 accumulators). x read once, cvt fp32->bf16 in
// flight. B tiles double-buffered (bcur/bnext) and next A block prefetched
// during the 12-WMMA burst so waits are loadcnt<=N, not 0.
// Outputs: q (row-major, pre-scaled 0.125), k (row-major), v TRANSPOSED.
// ---------------------------------------------------------------------------
__global__ __launch_bounds__(256) void qkv_kernel(const float* __restrict__ x,
                                                  const float* __restrict__ bq,
                                                  const float* __restrict__ bk,
                                                  const float* __restrict__ bv,
                                                  const __bf16* __restrict__ wswz,
                                                  __bf16* __restrict__ qg,
                                                  __bf16* __restrict__ kg,
                                                  __bf16* __restrict__ vtg) {
  const int lane = threadIdx.x & 31;
  const int wv   = threadIdx.x >> 5;
  const int l16  = lane & 15;
  const int half = lane >> 4;
  const int mt   = blockIdx.x * 8 + wv;   // 0..2047 row tiles
  const int m0   = mt << 4;
  const float* xr = x + (size_t)(m0 + l16) * H_DIM + half * 8;

  auto loadB = [&](int ks, int t) {
    const int mat = t >> 2, nt = t & 3;
    const __bf16* bp =
        wswz + ((size_t)((mat * KSTEPS + ks) * 4 + nt) * 32 + lane) * 16;
    V16 b;
    b.h[0] = *(const v8bf*)bp;
    b.h[1] = *(const v8bf*)(bp + 8);
    return b;
  };

  v8f acc[12];
#pragma unroll
  for (int t = 0; t < 12; ++t)
#pragma unroll
    for (int r = 0; r < 8; ++r) acc[t][r] = 0.0f;

  // prologue: first A block (floats) + first B tile in flight
  float4 fc[4];
  fc[0] = *(const float4*)(xr);
  fc[1] = *(const float4*)(xr + 4);
  fc[2] = *(const float4*)(xr + 16);
  fc[3] = *(const float4*)(xr + 20);
  V16 bcur = loadB(0, 0);

  for (int ks = 0; ks < KSTEPS; ++ks) {
    const float fa[16] = {fc[0].x, fc[0].y, fc[0].z, fc[0].w,
                          fc[1].x, fc[1].y, fc[1].z, fc[1].w,
                          fc[2].x, fc[2].y, fc[2].z, fc[2].w,
                          fc[3].x, fc[3].y, fc[3].z, fc[3].w};
    V16 a;
#pragma unroll
    for (int i = 0; i < 16; ++i) a.v[i] = (__bf16)fa[i];

    // issue next A block early; its latency hides under 12 WMMAs
    float4 fn[4];
    if (ks + 1 < KSTEPS) {
      const float* pn = xr + (ks + 1) * 32;
      fn[0] = *(const float4*)(pn);
      fn[1] = *(const float4*)(pn + 4);
      fn[2] = *(const float4*)(pn + 16);
      fn[3] = *(const float4*)(pn + 20);
    }
#pragma unroll
    for (int t = 0; t < 12; ++t) {
      V16 bnext = (t < 11) ? loadB(ks, t + 1) : loadB((ks + 1) % KSTEPS, 0);
      acc[t] = wmma_bf16(a.v, bcur.v, acc[t]);
      bcur = bnext;
    }
    if (ks + 1 < KSTEPS) {
      fc[0] = fn[0]; fc[1] = fn[1]; fc[2] = fn[2]; fc[3] = fn[3];
    }
  }

  // Epilogue: C/D layout -> lane holds column n=l16, VGPR r holds row r+8*half
#pragma unroll
  for (int t = 0; t < 12; ++t) {
    const int mat = t >> 2, nt = t & 3;
    const int n   = nt * 16 + l16;
    if (mat == 0) {
      const float bb = bq[n];
#pragma unroll
      for (int r = 0; r < 8; ++r) {
        const int row = m0 + half * 8 + r;
        qg[(size_t)row * D_DIM + n] = (__bf16)((acc[t][r] + bb) * 0.125f);
      }
    } else if (mat == 1) {
      const float bb = bk[n];
#pragma unroll
      for (int r = 0; r < 8; ++r) {
        const int row = m0 + half * 8 + r;
        kg[(size_t)row * D_DIM + n] = (__bf16)(acc[t][r] + bb);
      }
    } else {
      const float bb = bv[n];
      V8 pk;
#pragma unroll
      for (int r = 0; r < 8; ++r) pk.e[r] = (__bf16)(acc[t][r] + bb);
      __bf16* dst = vtg + ((size_t)(m0 >> 12) * D_DIM + n) * S_LEN +
                    (m0 & (S_LEN - 1)) + half * 8;
      *(v8bf*)dst = pk.v;
    }
  }
}

// ---------------------------------------------------------------------------
// Phase 2: causal flash attention. One wave = 32 query rows (2 row tiles),
// streams key blocks of 32: 16 WMMAs per block against 16 b128 loads.
// Phasing: K loads -> score WMMAs (K regs die) -> V loads issued -> softmax
// VALU hides V latency -> P through LDS -> P@V WMMAs. Mask only on the
// diagonal block (t0 == q0), wave-uniform.
// ---------------------------------------------------------------------------
__global__ __launch_bounds__(256) void fa_kernel(const __bf16* __restrict__ qg,
                                                 const __bf16* __restrict__ kg,
                                                 const __bf16* __restrict__ vtg,
                                                 float* __restrict__ out) {
  __shared__ __align__(16) __bf16 plds[8][32][32];
  const int lane  = threadIdx.x & 31;
  const int wv    = threadIdx.x >> 5;
  const int l16   = lane & 15;
  const int half  = lane >> 4;
  const int wg    = blockIdx.x * 8 + wv;  // 1024 waves total
  const int batch = wg >> 7;              // 128 q-tiles (of 32 rows) per batch
  const int q0    = (wg & 127) << 5;

  const __bf16* qb = qg + (size_t)batch * S_LEN * D_DIM;
  const __bf16* kb = kg + (size_t)batch * S_LEN * D_DIM;
  const __bf16* vt = vtg + (size_t)batch * D_DIM * S_LEN;
  float* ob        = out + (size_t)batch * S_LEN * D_DIM;

  // Q A-tiles: 2 row tiles x 2 d-halves (contiguous A layout)
  V16 aq[2][2];
#pragma unroll
  for (int rt = 0; rt < 2; ++rt) {
    const __bf16* qr = qb + (size_t)(q0 + rt * 16 + l16) * D_DIM + half * 16;
#pragma unroll
    for (int dh = 0; dh < 2; ++dh) {
      aq[rt][dh].h[0] = *(const v8bf*)(qr + dh * 32);
      aq[rt][dh].h[1] = *(const v8bf*)(qr + dh * 32 + 8);
    }
  }

  float mrow[2][8], lrow[2][8];
  v8f acc[2][4];
#pragma unroll
  for (int rt = 0; rt < 2; ++rt) {
#pragma unroll
    for (int r = 0; r < 8; ++r) { mrow[rt][r] = -1e30f; lrow[rt][r] = 0.0f; }
#pragma unroll
    for (int dt = 0; dt < 4; ++dt)
#pragma unroll
      for (int r = 0; r < 8; ++r) acc[rt][dt][r] = 0.0f;
  }

  for (int t0 = 0; t0 <= q0 + 31; t0 += 32) {
    // ---- K tiles (B-operand = rows of K, sequential K layout)
    V16 bk_[2][2];
#pragma unroll
    for (int kt = 0; kt < 2; ++kt) {
      const __bf16* kr = kb + (size_t)(t0 + kt * 16 + l16) * D_DIM + half * 16;
      bk_[kt][0].h[0] = *(const v8bf*)(kr);
      bk_[kt][0].h[1] = *(const v8bf*)(kr + 8);
      bk_[kt][1].h[0] = *(const v8bf*)(kr + 32);
      bk_[kt][1].h[1] = *(const v8bf*)(kr + 40);
    }
    __builtin_prefetch(kb + (size_t)(t0 + 32) * D_DIM + lane * 16, 0, 0);

    // ---- scores: S = Q @ K^T
    v8f s[2][2];
#pragma unroll
    for (int rt = 0; rt < 2; ++rt)
#pragma unroll
      for (int kt = 0; kt < 2; ++kt) {
        v8f z;
#pragma unroll
        for (int r = 0; r < 8; ++r) z[r] = 0.0f;
        z          = wmma_bf16(aq[rt][0].v, bk_[kt][0].v, z);
        s[rt][kt]  = wmma_bf16(aq[rt][1].v, bk_[kt][1].v, z);
      }

    // ---- issue V loads now; softmax VALU below hides their latency
    V16 bvt[4];
#pragma unroll
    for (int dt = 0; dt < 4; ++dt) {
      const __bf16* vr = vt + (size_t)(dt * 16 + l16) * S_LEN + t0 + half * 16;
      bvt[dt].h[0] = *(const v8bf*)(vr);
      bvt[dt].h[1] = *(const v8bf*)(vr + 8);
    }

    // ---- causal mask: only the diagonal block needs it (wave-uniform test)
    if (t0 == q0) {
#pragma unroll
      for (int rt = 0; rt < 2; ++rt)
#pragma unroll
        for (int kt = 0; kt < 2; ++kt) {
          const int tkey = t0 + kt * 16 + l16;
#pragma unroll
          for (int r = 0; r < 8; ++r) {
            const int qrow = q0 + rt * 16 + half * 8 + r;
            s[rt][kt][r] = (tkey > qrow) ? -1e30f : s[rt][kt][r];
          }
        }
    }

    // ---- online softmax (row = VGPR within a 16-lane half)
#pragma unroll
    for (int rt = 0; rt < 2; ++rt) {
      float scal[8];
#pragma unroll
      for (int r = 0; r < 8; ++r) {
        float p = fmaxf(s[rt][0][r], s[rt][1][r]);
        p = fmaxf(p, __shfl_xor(p, 1, 16));
        p = fmaxf(p, __shfl_xor(p, 2, 16));
        p = fmaxf(p, __shfl_xor(p, 4, 16));
        p = fmaxf(p, __shfl_xor(p, 8, 16));
        const float mn = fmaxf(mrow[rt][r], p);
        scal[r]     = __expf(mrow[rt][r] - mn);
        mrow[rt][r] = mn;
      }
#pragma unroll
      for (int kt = 0; kt < 2; ++kt)
#pragma unroll
        for (int r = 0; r < 8; ++r)
          s[rt][kt][r] = __expf(s[rt][kt][r] - mrow[rt][r]);
#pragma unroll
      for (int r = 0; r < 8; ++r) {
        float p = s[rt][0][r] + s[rt][1][r];
        p += __shfl_xor(p, 1, 16);
        p += __shfl_xor(p, 2, 16);
        p += __shfl_xor(p, 4, 16);
        p += __shfl_xor(p, 8, 16);
        lrow[rt][r] = lrow[rt][r] * scal[r] + p;
      }
#pragma unroll
      for (int dt = 0; dt < 4; ++dt)
#pragma unroll
        for (int r = 0; r < 8; ++r) acc[rt][dt][r] *= scal[r];

      // P: C-layout -> row-major LDS (wave-private; DS ops are in-order)
#pragma unroll
      for (int kt = 0; kt < 2; ++kt)
#pragma unroll
        for (int r = 0; r < 8; ++r)
          plds[wv][rt * 16 + half * 8 + r][kt * 16 + l16] =
              (__bf16)s[rt][kt][r];
    }
    asm volatile("" ::: "memory");

    // ---- O += P @ V  (B-operand = rows of V^T, contiguous)
#pragma unroll
    for (int rt = 0; rt < 2; ++rt) {
      V16 ap;
      ap.h[0] = *(const v8bf*)&plds[wv][rt * 16 + l16][half * 16];
      ap.h[1] = *(const v8bf*)&plds[wv][rt * 16 + l16][half * 16 + 8];
#pragma unroll
      for (int dt = 0; dt < 4; ++dt)
        acc[rt][dt] = wmma_bf16(ap.v, bvt[dt].v, acc[rt][dt]);
    }
  }

  // ---- epilogue: O / l
#pragma unroll
  for (int rt = 0; rt < 2; ++rt) {
    float inv[8];
#pragma unroll
    for (int r = 0; r < 8; ++r) inv[r] = 1.0f / lrow[rt][r];
#pragma unroll
    for (int dt = 0; dt < 4; ++dt)
#pragma unroll
      for (int r = 0; r < 8; ++r)
        ob[(size_t)(q0 + rt * 16 + half * 8 + r) * D_DIM + dt * 16 + l16] =
            acc[rt][dt][r] * inv[r];
  }
}

// ---------------------------------------------------------------------------
extern "C" void kernel_launch(void* const* d_in, const int* in_sizes, int n_in,
                              void* d_out, int out_size, void* d_ws,
                              size_t ws_size, hipStream_t stream) {
  (void)in_sizes; (void)n_in; (void)out_size; (void)ws_size;
  const float* x  = (const float*)d_in[0];
  const float* Wq = (const float*)d_in[1];
  const float* bq = (const float*)d_in[2];
  const float* Wk = (const float*)d_in[3];
  const float* bk = (const float*)d_in[4];
  const float* Wv = (const float*)d_in[5];
  const float* bv = (const float*)d_in[6];
  float* out = (float*)d_out;

  char* ws = (char*)d_ws;
  const size_t QKV_BYTES = (size_t)BATCH * S_LEN * D_DIM * sizeof(__bf16); // 4MB
  __bf16* wswz = (__bf16*)(ws);                       // 288 KB swizzled weights
  __bf16* qg   = (__bf16*)(ws + (1u << 19));
  __bf16* kg   = (__bf16*)(ws + (1u << 19) + QKV_BYTES);
  __bf16* vtg  = (__bf16*)(ws + (1u << 19) + 2 * QKV_BYTES);

  wprep_kernel<<<dim3(3 * KSTEPS * 4), dim3(32), 0, stream>>>(Wq, Wk, Wv, wswz);
  qkv_kernel<<<dim3((BATCH * S_LEN / 16) / 8), dim3(256), 0, stream>>>(
      x, bq, bk, bv, wswz, qg, kg, vtg);
  fa_kernel<<<dim3((BATCH * S_LEN / 32) / 8), dim3(256), 0, stream>>>(
      qg, kg, vtg, out);
}
